// TinyAttention_23974507446858
// MI455X (gfx1250) — matrix operand order repaired and checked by reading, not compiled, not run on hardware
//
#include <hip/hip_runtime.h>
#include <hip/hip_bf16.h>

typedef __attribute__((ext_vector_type(16))) __bf16       v16bf;
typedef __attribute__((ext_vector_type(8)))  float        v8f;
typedef __attribute__((ext_vector_type(4)))  unsigned int v4u;
typedef __attribute__((ext_vector_type(4)))  int          v4i;

union Frag16 { v16bf v; v4u q[2]; };
union Pack8  { __bf16 e[8]; v4u q; };

#if defined(__has_builtin)
#  if __has_builtin(__builtin_amdgcn_global_load_async_to_lds_b128)
#    define USE_ASYNC_LDS 1
#  endif
#  if __has_builtin(__builtin_amdgcn_cvt_pk_bf16_f32)
#    define HAVE_PK_BF16 1
#  endif
#endif
#ifndef USE_ASYNC_LDS
#  define USE_ASYNC_LDS 0
#endif
#ifndef HAVE_PK_BF16
#  define HAVE_PK_BF16 0
#endif

__device__ __forceinline__ __bf16 f2bf(float f) {
  union { float f; unsigned int u; } in; in.f = f;
  unsigned int u = in.u;
  u += 0x7FFFu + ((u >> 16) & 1u);          // round-to-nearest-even
  union { unsigned short s; __bf16 b; } out; out.s = (unsigned short)(u >> 16);
  return out.b;
}

// convert two f32 -> two packed bf16 at p.e[2*idx], single VALU op when available
__device__ __forceinline__ void pack2(Pack8& p, int idx, float a, float b) {
#if HAVE_PK_BF16
  auto r = __builtin_amdgcn_cvt_pk_bf16_f32(a, b);
  __builtin_memcpy(&p.e[idx * 2], &r, 4);
#else
  p.e[idx * 2] = f2bf(a);
  p.e[idx * 2 + 1] = f2bf(b);
#endif
}

// 16-byte global -> LDS staging copy (async DMA path on CDNA5, ASYNCcnt-tracked)
__device__ __forceinline__ void stage16(__bf16* lds_dst, const __bf16* g_src) {
#if USE_ASYNC_LDS
  __builtin_amdgcn_global_load_async_to_lds_b128(
      (__attribute__((address_space(1))) v4i*)g_src,
      (__attribute__((address_space(3))) v4i*)lds_dst, 0, 0);
#else
  *reinterpret_cast<v4u*>(lds_dst) = *reinterpret_cast<const v4u*>(g_src);
#endif
}

__device__ __forceinline__ void stage_wait() {
#if USE_ASYNC_LDS
#  if __has_builtin(__builtin_amdgcn_s_wait_asynccnt)
  __builtin_amdgcn_s_wait_asynccnt(0);
#  else
  asm volatile("s_wait_asynccnt 0x0" ::: "memory");
#  endif
#endif
  __syncthreads();
}

// A-fragment (16x32 bf16, MxK): kb = 8*(lane>>4); elems {kb..kb+7, kb+16..kb+23}.
__device__ __forceinline__ v16bf load_a_frag(const __bf16* row, int h) {
  Frag16 f;
  const int kb = h * 8;
  f.q[0] = *reinterpret_cast<const v4u*>(row + kb);
  f.q[1] = *reinterpret_cast<const v4u*>(row + kb + 16);
  return f.v;
}

// B-fragment (32x16 bf16, KxN): contraction-major col; kb = 8*(lane>>4); elems {kb..kb+7, kb+16..kb+23}.
__device__ __forceinline__ v16bf load_b_frag(const __bf16* col, int h) {
  Frag16 f;
  const int kb = h * 8;
  f.q[0] = *reinterpret_cast<const v4u*>(col + kb);
  f.q[1] = *reinterpret_cast<const v4u*>(col + kb + 16);
  return f.v;
}

// ------------- Kernel 0: convert + transpose weights to bf16 -------------
// WqkvT: [192][8192] bf16,  WffnT: [4096][64] bf16 (contraction-major, packed stores)
__global__ __launch_bounds__(256) void k_convert_w(
    const float* __restrict__ Wqkv, const float* __restrict__ Wffn,
    __bf16* __restrict__ WqkvT, __bf16* __restrict__ WffnT) {
  const int run = blockIdx.x * 256 + threadIdx.x;
  const int R1 = 192 * 1024;                     // 192 cols x 1024 runs of 8 k
  if (run < R1) {
    const int n = run >> 10, kc = (run & 1023) * 8;
    Pack8 p;
#pragma unroll
    for (int e = 0; e < 4; e++)
      pack2(p, e, Wqkv[(size_t)(kc + 2 * e) * 192 + n],
                  Wqkv[(size_t)(kc + 2 * e + 1) * 192 + n]);
    *reinterpret_cast<v4u*>(WqkvT + (size_t)n * 8192 + kc) = p.q;
  } else if (run - R1 < 4096 * 8) {
    const int r2 = run - R1;                     // 4096 cols x 8 runs of 8 k
    const int n = r2 >> 3, kc = (r2 & 7) * 8;
    Pack8 p;
#pragma unroll
    for (int e = 0; e < 4; e++)
      pack2(p, e, Wffn[(size_t)(kc + 2 * e) * 4096 + n],
                  Wffn[(size_t)(kc + 2 * e + 1) * 4096 + n]);
    *reinterpret_cast<v4u*>(WffnT + (size_t)n * 64 + kc) = p.q;
  }
}

// ------------- Kernel 1: qkv = x @ W_qkv + b_qkv (full N, double-buffered) ------
// x:(8192,8192)f32. Writes bf16: qs[tok][64] (pre-scaled d^-0.5), ks[tok][64],
// vT[b][64][2048].
__global__ __launch_bounds__(128) void k_qkv_gemm(
    const float* __restrict__ x, const __bf16* __restrict__ WqkvT,
    const float* __restrict__ bqkv, __bf16* __restrict__ qs,
    __bf16* __restrict__ ks, __bf16* __restrict__ vT) {
  __shared__ __align__(16) __bf16 ldsA[2][64 * 64];    // [m][k]  2 x  8KB
  __shared__ __align__(16) __bf16 ldsB[2][192 * 64];   // [n][k]  2 x 24KB
  const int tid = threadIdx.x, wave = tid >> 5, lane = tid & 31;
  const int h = lane >> 4, nl = lane & 15;
  const int m0 = blockIdx.x * 64;
  const v8f vzero = {};
  v8f acc[12];
#pragma unroll
  for (int t = 0; t < 12; t++) acc[t] = vzero;

  // ---- prologue: stage k0 = 0 into buffer 0 ----
#pragma unroll
  for (int jj = 0; jj < 12; jj++) {
    const int run = tid + jj * 128;
    const int n = run >> 3, kc = (run & 7) * 8;
    stage16(&ldsB[0][n * 64 + kc], WqkvT + (size_t)n * 8192 + kc);
  }
#pragma unroll
  for (int jj = 0; jj < 4; jj++) {
    const int run = tid + jj * 128;
    const int m = run >> 3, kc = (run & 7) * 8;
    const float* g = x + (size_t)(m0 + m) * 8192 + kc;
    const float4 f0 = *reinterpret_cast<const float4*>(g);
    const float4 f1 = *reinterpret_cast<const float4*>(g + 4);
    Pack8 p;
    pack2(p, 0, f0.x, f0.y); pack2(p, 1, f0.z, f0.w);
    pack2(p, 2, f1.x, f1.y); pack2(p, 3, f1.z, f1.w);
    *reinterpret_cast<v4u*>(&ldsA[0][m * 64 + kc]) = p.q;
  }

  for (int it = 0; it < 128; ++it) {
    const int cur = it & 1, nxt = cur ^ 1;
    stage_wait();                                  // buf[cur] ready; buf[nxt] free
    const bool has_next = (it + 1) < 128;
    const int k0n = (it + 1) * 64;
    float4 pf[8];
    if (has_next) {
      // issue next B tile (async DMA to idle buffer) and A global loads now;
      // both overlap with the WMMA stream below
#pragma unroll
      for (int jj = 0; jj < 12; jj++) {
        const int run = tid + jj * 128;
        const int n = run >> 3, kc = (run & 7) * 8;
        stage16(&ldsB[nxt][n * 64 + kc], WqkvT + (size_t)n * 8192 + k0n + kc);
      }
#pragma unroll
      for (int jj = 0; jj < 4; jj++) {
        const int run = tid + jj * 128;
        const int m = run >> 3, kc = (run & 7) * 8;
        const float* g = x + (size_t)(m0 + m) * 8192 + k0n + kc;
        pf[2 * jj]     = *reinterpret_cast<const float4*>(g);
        pf[2 * jj + 1] = *reinterpret_cast<const float4*>(g + 4);
      }
    }
    // ---- compute on buf[cur]: 24 WMMAs ----
#pragma unroll
    for (int s = 0; s < 2; s++) {
      v16bf a = load_a_frag(&ldsA[cur][(wave * 16 + nl) * 64 + s * 32], h);
#pragma unroll
      for (int t = 0; t < 12; t++) {
        v16bf bfr = load_b_frag(&ldsB[cur][(t * 16 + nl) * 64 + s * 32], h);
        acc[t] = __builtin_amdgcn_wmma_f32_16x16x32_bf16(
            false, a, false, bfr, (short)0, acc[t], false, false);
      }
    }
    // ---- convert + store prefetched A into idle buffer ----
    if (has_next) {
#pragma unroll
      for (int jj = 0; jj < 4; jj++) {
        const int run = tid + jj * 128;
        const int m = run >> 3, kc = (run & 7) * 8;
        const float4 f0 = pf[2 * jj], f1 = pf[2 * jj + 1];
        Pack8 p;
        pack2(p, 0, f0.x, f0.y); pack2(p, 1, f0.z, f0.w);
        pack2(p, 2, f1.x, f1.y); pack2(p, 3, f1.z, f1.w);
        *reinterpret_cast<v4u*>(&ldsA[nxt][m * 64 + kc]) = p.q;
      }
    }
  }
#pragma unroll
  for (int t = 0; t < 12; t++) {
    const int col = t * 16 + nl;                   // 0..191
    const float bias = bqkv[col];
#pragma unroll
    for (int i = 0; i < 8; i++) {
      const int tok = m0 + wave * 16 + 8 * h + i;
      const float val = acc[t][i] + bias;
      if (t < 4) {                                 // q, pre-scaled
        qs[(size_t)tok * 64 + col] = f2bf(val * 0.125f);
      } else if (t < 8) {                          // k
        ks[(size_t)tok * 64 + (col - 64)] = f2bf(val);
      } else {                                     // v, transposed [b][d][s]
        const int d = col - 128, bb = tok >> 11, ss = tok & 2047;
        vT[((size_t)bb * 64 + d) * 2048 + ss] = f2bf(val);
      }
    }
  }
}

// ------------- Kernel 2: causal flash attention (all-bf16 staging) -------------
__global__ __launch_bounds__(128) void k_attention(
    const __bf16* __restrict__ qs, const __bf16* __restrict__ ks,
    const __bf16* __restrict__ vT, __bf16* __restrict__ attnb) {
  __shared__ __align__(16) __bf16 ldsQ[64 * 64];    // [m][d]
  __shared__ __align__(16) __bf16 ldsK[64 * 64];    // [key][d]
  __shared__ __align__(16) __bf16 ldsVT[64 * 64];   // [d][key]
  __shared__ __align__(16) __bf16 ldsP[4 * 16 * 64];
  const int tid = threadIdx.x, wave = tid >> 5, lane = tid & 31;
  const int h = lane >> 4, nl = lane & 15;
  const int qt = blockIdx.x, b = blockIdx.y;
  const int tok0 = b * 2048, qi0 = qt * 64;
  const v8f vzero = {};

#pragma unroll
  for (int jj = 0; jj < 4; jj++) {                  // stage Q (already scaled)
    const int run = tid + jj * 128;
    const int r = run >> 3, dc = (run & 7) * 8;
    stage16(&ldsQ[r * 64 + dc], qs + (size_t)(tok0 + qi0 + r) * 64 + dc);
  }
  v8f oacc[4];
#pragma unroll
  for (int t = 0; t < 4; t++) oacc[t] = vzero;
  float mrow[8], lrow[8];
#pragma unroll
  for (int i = 0; i < 8; i++) { mrow[i] = -3.0e38f; lrow[i] = 0.0f; }

  for (int j = 0; j <= qt; j++) {
    __syncthreads();
#pragma unroll
    for (int jj = 0; jj < 4; jj++) {                // K: [key][d] copy
      const int run = tid + jj * 128;
      const int r = run >> 3, dc = (run & 7) * 8;
      stage16(&ldsK[r * 64 + dc], ks + (size_t)(tok0 + j * 64 + r) * 64 + dc);
    }
#pragma unroll
    for (int jj = 0; jj < 4; jj++) {                // V^T: [d][key] copy
      const int run = tid + jj * 128;
      const int d = run >> 3, kc = (run & 7) * 8;
      stage16(&ldsVT[d * 64 + kc], vT + ((size_t)b * 64 + d) * 2048 + j * 64 + kc);
    }
    stage_wait();

    v8f sc[4];
#pragma unroll
    for (int t = 0; t < 4; t++) sc[t] = vzero;
#pragma unroll
    for (int s = 0; s < 2; s++) {
      v16bf a = load_a_frag(&ldsQ[(wave * 16 + nl) * 64 + s * 32], h);
#pragma unroll
      for (int t = 0; t < 4; t++) {
        v16bf bk = load_b_frag(&ldsK[(t * 16 + nl) * 64 + s * 32], h);
        sc[t] = __builtin_amdgcn_wmma_f32_16x16x32_bf16(
            false, a, false, bk, (short)0, sc[t], false, false);
      }
    }
#pragma unroll
    for (int i = 0; i < 8; i++) {
      const int qrow = qi0 + wave * 16 + 8 * h + i;
      float rm = -3.0e38f;
#pragma unroll
      for (int t = 0; t < 4; t++) {
        const int key = j * 64 + t * 16 + nl;
        float s = sc[t][i];
        if (key > qrow) s = -10000.0f;
        sc[t][i] = s;
        rm = fmaxf(rm, s);
      }
#pragma unroll
      for (int off = 1; off < 16; off <<= 1)
        rm = fmaxf(rm, __shfl_xor(rm, off, 32));
      const float mnew = fmaxf(mrow[i], rm);
      const float corr = __expf(mrow[i] - mnew);
      mrow[i] = mnew;
      float ps = 0.0f;
#pragma unroll
      for (int t = 0; t < 4; t++) {
        float p = __expf(sc[t][i] - mnew);
        sc[t][i] = p;
        ps += p;
      }
#pragma unroll
      for (int off = 1; off < 16; off <<= 1)
        ps += __shfl_xor(ps, off, 32);
      lrow[i] = lrow[i] * corr + ps;
#pragma unroll
      for (int t = 0; t < 4; t++) oacc[t][i] *= corr;
    }
#pragma unroll
    for (int t = 0; t < 4; t++)
#pragma unroll
      for (int i = 0; i < 8; i++)
        ldsP[wave * 1024 + (8 * h + i) * 64 + t * 16 + nl] = f2bf(sc[t][i]);
    __syncthreads();
#pragma unroll
    for (int s = 0; s < 2; s++) {
      v16bf a = load_a_frag(&ldsP[wave * 1024 + nl * 64 + s * 32], h);
#pragma unroll
      for (int t = 0; t < 4; t++) {
        v16bf bv = load_b_frag(&ldsVT[(t * 16 + nl) * 64 + s * 32], h);
        oacc[t] = __builtin_amdgcn_wmma_f32_16x16x32_bf16(
            false, a, false, bv, (short)0, oacc[t], false, false);
      }
    }
  }
#pragma unroll
  for (int i = 0; i < 8; i++) {
    const float inv = 1.0f / lrow[i];
    const int tok = tok0 + qi0 + wave * 16 + 8 * h + i;
#pragma unroll
    for (int t = 0; t < 4; t++)
      attnb[(size_t)tok * 64 + t * 16 + nl] = f2bf(oacc[t][i] * inv);
  }
}

// ------------- Kernel 3: out = attn @ W_ffn + b_ffn -------------
__global__ __launch_bounds__(128) void k_ffn_gemm(
    const __bf16* __restrict__ attnb, const __bf16* __restrict__ WffnT,
    const float* __restrict__ bffn, float* __restrict__ out) {
  __shared__ __align__(16) __bf16 ldsA[64 * 64];    // [m][k]
  __shared__ __align__(16) __bf16 ldsB[64 * 64];    // [n][k]
  const int tid = threadIdx.x, wave = tid >> 5, lane = tid & 31;
  const int h = lane >> 4, nl = lane & 15;
  const int m0 = blockIdx.y * 64, n0 = blockIdx.x * 64;
  const v8f vzero = {};
#pragma unroll
  for (int jj = 0; jj < 4; jj++) {
    const int run = tid + jj * 128;
    const int r = run >> 3, kc = (run & 7) * 8;
    stage16(&ldsA[r * 64 + kc], attnb + (size_t)(m0 + r) * 64 + kc);
    stage16(&ldsB[r * 64 + kc], WffnT + (size_t)(n0 + r) * 64 + kc);
  }
  stage_wait();
  v8f acc[4];
#pragma unroll
  for (int t = 0; t < 4; t++) acc[t] = vzero;
#pragma unroll
  for (int s = 0; s < 2; s++) {
    v16bf a = load_a_frag(&ldsA[(wave * 16 + nl) * 64 + s * 32], h);
#pragma unroll
    for (int t = 0; t < 4; t++) {
      v16bf bw = load_b_frag(&ldsB[(t * 16 + nl) * 64 + s * 32], h);
      acc[t] = __builtin_amdgcn_wmma_f32_16x16x32_bf16(
          false, a, false, bw, (short)0, acc[t], false, false);
    }
  }
#pragma unroll
  for (int t = 0; t < 4; t++) {
    const int col = n0 + t * 16 + nl;
    const float bias = bffn[col];
#pragma unroll
    for (int i = 0; i < 8; i++) {
      const int row = m0 + wave * 16 + 8 * h + i;
      out[(size_t)row * 4096 + col] = acc[t][i] + bias;
    }
  }
}

extern "C" void kernel_launch(void* const* d_in, const int* in_sizes, int n_in,
                              void* d_out, int out_size, void* d_ws, size_t ws_size,
                              hipStream_t stream) {
  const float* x    = (const float*)d_in[0];
  // d_in[1] = attention_mask (bool triu, k=1): applied analytically in-kernel
  const float* Wqkv = (const float*)d_in[2];
  const float* bqkv = (const float*)d_in[3];
  const float* Wffn = (const float*)d_in[4];
  const float* bffn = (const float*)d_in[5];
  float* out = (float*)d_out;

  char* ws = (char*)d_ws;                            // all bf16, 16B-aligned
  __bf16* qs    = (__bf16*)(ws + 0);                 // 8192*64   = 1 MB
  __bf16* ks    = (__bf16*)(ws + (1u << 20));        // 8192*64   = 1 MB
  __bf16* vT    = (__bf16*)(ws + (2u << 20));        // 4*64*2048 = 1 MB
  __bf16* attnb = (__bf16*)(ws + (3u << 20));        // 8192*64   = 1 MB
  __bf16* WqkvT = (__bf16*)(ws + (4u << 20));        // 192*8192  = 3 MB
  __bf16* WffnT = (__bf16*)(ws + (7u << 20));        // 4096*64   = 0.5 MB

  const int convRuns = 192 * 1024 + 4096 * 8;
  k_convert_w<<<dim3((convRuns + 255) / 256), 256, 0, stream>>>(Wqkv, Wffn, WqkvT, WffnT);
  k_qkv_gemm <<<dim3(128),      128, 0, stream>>>(x, WqkvT, bqkv, qs, ks, vT);
  k_attention<<<dim3(32, 4),    128, 0, stream>>>(qs, ks, vT, attnb);
  k_ffn_gemm <<<dim3(64, 128),  128, 0, stream>>>(attnb, WffnT, bffn, out);
}
